// Experts_48868137894561
// MI455X (gfx1250) — matrix operand order repaired and checked by reading, not compiled, not run on hardware
//
#include <hip/hip_runtime.h>
#include <hip/hip_bf16.h>

// Problem constants (from reference)
#define B_ROWS 8192
#define E_NUM  8
#define F1     1024
#define F2     2048
#define C_NUM  100
#define TM     64                 // rows per block tile
#define TILES_PER_E (B_ROWS / TM) // 128
#define NTHREADS 512              // 16 wave32 waves

typedef __attribute__((ext_vector_type(16))) _Float16 v16h;
typedef __attribute__((ext_vector_type(8)))  _Float16 v8h;
typedef __attribute__((ext_vector_type(8)))  float    v8f;

// LDS strides (in halves): rows 16B-aligned, bank-conflict-spread
#define XS_STRIDE  1032   // 1024 + 8 ; 2064 B/row = 16*129
#define HS_STRIDE  136    // 128 + 8  ; 272 B/row
#define W1T_STRIDE 40     // 32 + 8   ; transposed W1 slab rows (n-major)
#define W1T_SLAB   (128 * W1T_STRIDE)

#if __has_builtin(__builtin_amdgcn_sched_group_barrier)
#define SGB(mask, cnt) __builtin_amdgcn_sched_group_barrier(mask, cnt, 0)
#else
#define SGB(mask, cnt)
#endif

// ---------------- bucketing kernels ----------------
__global__ void zero_counts_k(int* counts) {
    if (threadIdx.x < E_NUM) counts[threadIdx.x] = 0;
}

__global__ void count_rows_k(const int* __restrict__ domain, int* __restrict__ counts) {
    int i = blockIdx.x * blockDim.x + threadIdx.x;
    if (i < B_ROWS) atomicAdd(&counts[domain[i]], 1);
}

__global__ void scan_counts_k(const int* __restrict__ counts, int* __restrict__ offsets,
                              int* __restrict__ cursors) {
    if (threadIdx.x == 0) {
        int acc = 0;
        for (int e = 0; e < E_NUM; ++e) { offsets[e] = acc; cursors[e] = acc; acc += counts[e]; }
    }
}

__global__ void scatter_rows_k(const int* __restrict__ domain, int* __restrict__ cursors,
                               int* __restrict__ rowlist) {
    int i = blockIdx.x * blockDim.x + threadIdx.x;
    if (i < B_ROWS) {
        int p = atomicAdd(&cursors[domain[i]], 1);
        rowlist[p] = i;
    }
}

// ---------------- fused expert MLP + softmax ----------------
__global__ __launch_bounds__(NTHREADS, 1)
void moe_fused_k(const int* __restrict__ rowlist, const int* __restrict__ counts,
                 const int* __restrict__ offsets,
                 const float* __restrict__ x,  const float* __restrict__ W1,
                 const float* __restrict__ b1, const float* __restrict__ W2,
                 const float* __restrict__ b2, float* __restrict__ out)
{
    __shared__ __align__(16) _Float16 XS[TM * XS_STRIDE];   // 64 x 1024 f16 acts   (129 KB)
    __shared__ __align__(16) _Float16 HS[TM * HS_STRIDE];   // 64 x 128  f16 hidden (17 KB)
    __shared__ __align__(16) _Float16 W1T[2 * W1T_SLAB];    // double-buffered transposed slab (20 KB)
    __shared__ int ridx[TM];

    const int e   = blockIdx.x / TILES_PER_E;
    const int t   = blockIdx.x % TILES_PER_E;
    const int cnt = counts[e];
    if (t * TM >= cnt) return;
    const int off    = offsets[e];
    const int nvalid = min(TM, cnt - t * TM);

    const int tid  = threadIdx.x;
    const int lane = tid & 31;
    const int wave = tid >> 5;        // 16 waves
    const int lhid = lane & 15;       // column within 16-wide tile
    const int lgrp = lane >> 4;       // 0 or 1 (K-half selector)

    // GEMM1 tile mapping: N-tile = wave&7, M-tiles = (wave>>3)*2 + {0,1}
    const int nt1 = wave & 7;
    const int mb1 = (wave >> 3) << 1;
    // GEMM2 tile mapping (waves 0..13): N-tile = wave%7, M-tiles = (wave/7)*2 + {0,1}
    const int nt2 = wave % 7;
    const int mb2 = (wave / 7) << 1;

    if (tid < TM) {
        int r = t * TM + ((tid < nvalid) ? tid : 0);  // pad with first (valid) row
        ridx[tid] = rowlist[off + r];
    }
    __syncthreads();

    // Gather 64 rows of x, convert f32 -> f16 into LDS
    for (int i = tid; i < TM * (F1 / 4); i += NTHREADS) {
        int row = i >> 8;           // 256 float4 per row
        int c4  = i & 255;
        float4 v = reinterpret_cast<const float4*>(x + (size_t)ridx[row] * F1)[c4];
        int o = row * XS_STRIDE + c4 * 4;
        XS[o + 0] = (_Float16)v.x; XS[o + 1] = (_Float16)v.y;
        XS[o + 2] = (_Float16)v.z; XS[o + 3] = (_Float16)v.w;
    }

    v8f lacc[2] = {v8f{}, v8f{}};     // logits accumulators (waves 0..13)

    const size_t w1base = (size_t)e * F1 * F2;
    const size_t w2base = (size_t)e * F2 * C_NUM;

    for (int chunk = 0; chunk < F2 / 128; ++chunk) {
        const int nb = chunk * 128;
        v8f hacc[2] = {v8f{}, v8f{}};

        // Register buffer for the W1 k-slab: this thread owns
        // k-rows {wave + 16j}, n-quad = lane*4 within the 128-wide chunk.
        const float* gW = W1 + w1base + nb + (size_t)wave * F2 + (lane << 2);
        float4 st[2];
        #pragma unroll
        for (int j = 0; j < 2; ++j)
            st[j] = *reinterpret_cast<const float4*>(gW + (size_t)(16 * j) * F2);

        // ---- GEMM1: H[64x128] = X[64x1024] * W1[e][:, nb..nb+128), K tiled by 32 ----
        // Double-buffered W1T: one barrier per K-step.
        for (int kc = 0; kc < F1; kc += 32) {
            _Float16* slab = &W1T[((kc >> 5) & 1) * W1T_SLAB];
            // store staged slab TRANSPOSED: slab[n][k]
            #pragma unroll
            for (int j = 0; j < 2; ++j) {
                const int kcol = wave + 16 * j;
                const int nrow = lane << 2;
                slab[(nrow + 0) * W1T_STRIDE + kcol] = (_Float16)st[j].x;
                slab[(nrow + 1) * W1T_STRIDE + kcol] = (_Float16)st[j].y;
                slab[(nrow + 2) * W1T_STRIDE + kcol] = (_Float16)st[j].z;
                slab[(nrow + 3) * W1T_STRIDE + kcol] = (_Float16)st[j].w;
            }
            // preload next k-slab (branch-free wrap on last iteration; extra loads harmless)
            {
                const int kn = (kc + 32) & (F1 - 1);
                const float* gn = gW + (size_t)kn * F2;
                #pragma unroll
                for (int j = 0; j < 2; ++j)
                    st[j] = *reinterpret_cast<const float4*>(gn + (size_t)(16 * j) * F2);
                __builtin_prefetch(gn + (size_t)32 * F2, 0, 1);  // slab after next -> global_prefetch
            }
            __syncthreads();   // slab visible; prior-parity readers all past this point

            // B fragment (wave's N tile): element i needs K = lgrp*16 + i at fixed n
            //  -> two contiguous b128 loads from the transposed slab
            const _Float16* bp = &slab[((nt1 << 4) + lhid) * W1T_STRIDE + (lgrp << 4)];
            v8h b0 = *reinterpret_cast<const v8h*>(bp);
            v8h b1v = *reinterpret_cast<const v8h*>(bp + 8);
            v16h bfrag = __builtin_shufflevector(b0, b1v, 0,1,2,3,4,5,6,7,8,9,10,11,12,13,14,15);

            v16h af[2];
            #pragma unroll
            for (int q = 0; q < 2; ++q) {
                const _Float16* ap = &XS[(((mb1 + q) << 4) + lhid) * XS_STRIDE + kc + (lgrp << 4)];
                v8h a0 = *reinterpret_cast<const v8h*>(ap);
                v8h a1 = *reinterpret_cast<const v8h*>(ap + 8);
                af[q] = __builtin_shufflevector(a0, a1, 0,1,2,3,4,5,6,7,8,9,10,11,12,13,14,15);
            }
            #pragma unroll
            for (int q = 0; q < 2; ++q)
                hacc[q] = __builtin_amdgcn_wmma_f32_16x16x32_f16(
                    false, af[q], false, bfrag, (short)0, hacc[q], false, false);

            // Scheduling: issue all 6 LDS b128 reads, then both WMMAs back-to-back
            SGB(0x100, 6);
            SGB(0x008, 2);
        }

        // ---- bias + ReLU, write H chunk to LDS as f16 ----
        // (in-loop barriers already order prev chunk's GEMM2 HS reads before these writes)
        const float bv = b1[(size_t)e * F2 + nb + (nt1 << 4) + lhid];
        #pragma unroll
        for (int q = 0; q < 2; ++q) {
            #pragma unroll
            for (int r = 0; r < 8; ++r) {
                float vv = hacc[q][r] + bv;
                vv = vv > 0.f ? vv : 0.f;
                HS[(((mb1 + q) << 4) + (lgrp << 3) + r) * HS_STRIDE + (nt1 << 4) + lhid] = (_Float16)vv;
            }
        }
        __syncthreads();

        // ---- GEMM2: logits[64x112] += H[64x128] * W2[e][nb..nb+128, :] ----
        if (wave < 14) {
            const int  ccol = (nt2 << 4) + lhid;
            const bool cok  = ccol < C_NUM;
            #pragma unroll
            for (int kc2 = 0; kc2 < 128; kc2 += 32) {
                v16h b2frag;
                const float* wp = W2 + w2base + (size_t)(nb + kc2 + (lgrp << 4)) * C_NUM + ccol;
                #pragma unroll
                for (int i = 0; i < 16; ++i)
                    b2frag[i] = cok ? (_Float16)wp[i * C_NUM] : (_Float16)0.f;

                v16h hf[2];
                #pragma unroll
                for (int q = 0; q < 2; ++q) {
                    const _Float16* hp = &HS[(((mb2 + q) << 4) + lhid) * HS_STRIDE + kc2 + (lgrp << 4)];
                    v8h h0 = *reinterpret_cast<const v8h*>(hp);
                    v8h h1 = *reinterpret_cast<const v8h*>(hp + 8);
                    hf[q] = __builtin_shufflevector(h0, h1, 0,1,2,3,4,5,6,7,8,9,10,11,12,13,14,15);
                }
                #pragma unroll
                for (int q = 0; q < 2; ++q)
                    lacc[q] = __builtin_amdgcn_wmma_f32_16x16x32_f16(
                        false, hf[q], false, b2frag, (short)0, lacc[q], false, false);

                // Scheduling: 16 global reads, 4 LDS reads, then both WMMAs
                SGB(0x020, 16);
                SGB(0x100, 4);
                SGB(0x008, 2);
            }
        }
    }

    // ---- dump logits to LDS (reuse XS region), per-row softmax, scatter out ----
    float* LG = reinterpret_cast<float*>(XS);   // 64 x 116 f32 (29.7 KB < XS)
    __syncthreads();                            // XS no longer read as f16
    if (wave < 14) {
        #pragma unroll
        for (int q = 0; q < 2; ++q)
            #pragma unroll
            for (int r = 0; r < 8; ++r)
                LG[(((mb2 + q) << 4) + (lgrp << 3) + r) * 116 + (nt2 << 4) + lhid] = lacc[q][r];
    }
    __syncthreads();

    if (tid < nvalid) {
        const int    orow = ridx[tid];
        const float* bp2  = b2 + (size_t)e * C_NUM;
        float mx = -1e30f;
        for (int c = 0; c < C_NUM; ++c) {
            float v = LG[tid * 116 + c] + bp2[c];
            LG[tid * 116 + c] = v;
            mx = fmaxf(mx, v);
        }
        float s = 0.f;
        for (int c = 0; c < C_NUM; ++c) {
            float ev = __expf(LG[tid * 116 + c] - mx);
            LG[tid * 116 + c] = ev;
            s += ev;
        }
        float inv = 1.f / s;
        float* op = out + (size_t)orow * C_NUM;
        for (int c = 0; c < C_NUM; ++c) op[c] = LG[tid * 116 + c] * inv;
    }
}

// ---------------- launch ----------------
extern "C" void kernel_launch(void* const* d_in, const int* in_sizes, int n_in,
                              void* d_out, int out_size, void* d_ws, size_t ws_size,
                              hipStream_t stream) {
    const int*   domain = (const int*)  d_in[0];
    const float* x      = (const float*)d_in[1];
    const float* W1     = (const float*)d_in[2];
    const float* b1     = (const float*)d_in[3];
    const float* W2     = (const float*)d_in[4];
    const float* b2     = (const float*)d_in[5];
    float*       out    = (float*)d_out;

    int* ws      = (int*)d_ws;
    int* counts  = ws;           // [8]
    int* offsets = ws + 8;       // [8]
    int* cursors = ws + 16;      // [8]
    int* rowlist = ws + 24;      // [8192]

    zero_counts_k <<<1, 32, 0, stream>>>(counts);
    count_rows_k  <<<B_ROWS / 256, 256, 0, stream>>>(domain, counts);
    scan_counts_k <<<1, 1, 0, stream>>>(counts, offsets, cursors);
    scatter_rows_k<<<B_ROWS / 256, 256, 0, stream>>>(domain, cursors, rowlist);

    moe_fused_k<<<E_NUM * TILES_PER_E, NTHREADS, 0, stream>>>(
        rowlist, counts, offsets, x, W1, b1, W2, b2, out);
}
